// MemoryAttention_678604832955
// MI455X (gfx1250) — matrix operand order repaired and checked by reading, not compiled, not run on hardware
//
#include <hip/hip_runtime.h>
#include <hip/hip_bf16.h>
#include <stdint.h>

// ---------------------------------------------------------------------------
// MemoryAttention for MI455X (gfx1250): bf16 WMMA everywhere, flash softmax,
// bit-packed mask, 32-query wave tiles, cooperative async K/V LDS staging.
// B=4, NQ=1024, NM=4096, QD=MD=1024, H=16, HD=64, SCALE=1/8.
// ---------------------------------------------------------------------------

typedef __bf16 bf16;
typedef __attribute__((ext_vector_type(16))) __bf16 v16bf;
typedef __attribute__((ext_vector_type(8)))  __bf16 v8bf;
typedef __attribute__((ext_vector_type(8)))  float  v8f;

#define B_   4
#define NQ_  1024
#define NM_  4096
#define QD_  1024
#define H_   16
#define HD_  64
#define SCALE_ 0.125f
#define LOG2E_ 1.44269504088896340736f
#define MT_  2    // query m-tiles (of 16 rows) per wave
#define PAD_ 8    // LDS row padding (bf16 elements)

// ---- optional gfx1250 async global->LDS copy -------------------------------
#if defined(__has_builtin)
#if __has_builtin(__builtin_amdgcn_global_load_async_to_lds_b128)
#define HAVE_ASYNC_LDS 1
#endif
#endif

#ifdef HAVE_ASYNC_LDS
typedef int v4i_ __attribute__((__vector_size__(4 * sizeof(int))));
typedef __attribute__((address_space(1))) v4i_* as1_v4i_p;
typedef __attribute__((address_space(3))) v4i_* as3_v4i_p;

__device__ inline void async_copy16(const bf16* gp, bf16* lp) {
    // integer-cast address-space conversion (generic low 32 bits == LDS offset)
    __builtin_amdgcn_global_load_async_to_lds_b128(
        (as1_v4i_p)(uintptr_t)gp, (as3_v4i_p)(uintptr_t)lp, 0, 0);
}
__device__ inline void async_wait_all() {
#if __has_builtin(__builtin_amdgcn_s_wait_asynccnt)
    __builtin_amdgcn_s_wait_asynccnt(0);
#else
    asm volatile("s_wait_asynccnt 0x0" ::: "memory");
#endif
}
#endif

// ---- WMMA fragment loaders (per CDNA5 ISA 7.12.2 VGPR layouts, wave32) ----

// A-matrix 16x32 bf16, row-major source with leading dim `lda` (elements).
__device__ inline v16bf load_a_frag(const bf16* Ap, int lda) {
    const int l = threadIdx.x & 31;
    const int r = l & 15;
    const int s = l >> 4;
    const bf16* p = Ap + (size_t)r * lda + 8 * s;
    v8bf lo = *reinterpret_cast<const v8bf*>(p);
    v8bf hi = *reinterpret_cast<const v8bf*>(p + 16);
    v16bf a;
#pragma unroll
    for (int i = 0; i < 8; ++i) { a[i] = lo[i]; a[8 + i] = hi[i]; }
    return a;
}

// B-matrix 32x16 bf16 given as Bt[n][k] row-major (leading dim ldb).
__device__ inline v16bf load_b_frag(const bf16* Btp, int ldb) {
    const int l = threadIdx.x & 31;
    const int n = l & 15;
    const int h = l >> 4;
    const bf16* p = Btp + (size_t)n * ldb + 8 * h;
    v8bf lo = *reinterpret_cast<const v8bf*>(p);
    v8bf hi = *reinterpret_cast<const v8bf*>(p + 16);
    v16bf b;
#pragma unroll
    for (int i = 0; i < 8; ++i) { b[i] = lo[i]; b[8 + i] = hi[i]; }
    return b;
}

__device__ inline v8f wmma_bf16(v16bf a, v16bf b, v8f c) {
    return __builtin_amdgcn_wmma_f32_16x16x32_bf16(
        false, a, false, b, (short)0, c, false, false);
}

__device__ inline float half_max(float x) {
#pragma unroll
    for (int m = 1; m < 16; m <<= 1) x = fmaxf(x, __shfl_xor(x, m, 32));
    return x;
}
__device__ inline float half_sum(float x) {
#pragma unroll
    for (int m = 1; m < 16; m <<= 1) x += __shfl_xor(x, m, 32);
    return x;
}

// ---------------------------------------------------------------------------
__global__ void cvt_f32_bf16(const float* __restrict__ s, bf16* __restrict__ d,
                             long long n) {
    long long i = (long long)blockIdx.x * blockDim.x + threadIdx.x;
    long long stride = (long long)gridDim.x * blockDim.x;
    for (; i < n; i += stride) d[i] = (bf16)s[i];
}

__global__ void pack_mask_kernel(const int* __restrict__ m,
                                 uint32_t* __restrict__ mp, long long n) {
    long long i = (long long)blockIdx.x * blockDim.x + threadIdx.x;
    const long long stride = (long long)gridDim.x * blockDim.x;
    for (; i < n; i += stride) {
        unsigned long long bal = __ballot(m[i] != 0);
        if ((threadIdx.x & 31) == 0) mp[i >> 5] = (uint32_t)bal;
    }
}

// ---------------------------------------------------------------------------
// Generic bf16 GEMM:  out = scale * (A(MxK) @ W(NxK)^T + bias)
// MODE 0: bf16 row-major; MODE 1: bf16 transposed-per-batch (V^T);
// MODE 2: fp32 row-major.  Wave tile 32x64, K-step 32.
// ---------------------------------------------------------------------------
template <int MODE>
__global__ __launch_bounds__(256) void gemm_bf16_kernel(
    const bf16* __restrict__ A, const bf16* __restrict__ Bt,
    const float* __restrict__ bias, void* __restrict__ outp,
    int M, int N, int K, float scale, int rows_per_batch,
    long long out_batch_stride) {
    const int wave = blockIdx.x * (blockDim.x >> 5) + (threadIdx.x >> 5);
    const int ntiles = N >> 6;
    const int mt = wave / ntiles;
    const int nt = wave % ntiles;
    if (mt * 32 >= M) return;
    const int m0 = mt * 32, n0 = nt * 64;

    v8f acc[2][4];
#pragma unroll
    for (int i = 0; i < 2; ++i)
#pragma unroll
        for (int j = 0; j < 4; ++j) acc[i][j] = {};

    for (int k0 = 0; k0 < K; k0 += 32) {
        v16bf a0 = load_a_frag(A + (size_t)m0 * K + k0, K);
        v16bf a1 = load_a_frag(A + (size_t)(m0 + 16) * K + k0, K);
#pragma unroll
        for (int j = 0; j < 4; ++j) {
            v16bf bb = load_b_frag(Bt + (size_t)(n0 + j * 16) * K + k0, K);
            acc[0][j] = wmma_bf16(a0, bb, acc[0][j]);
            acc[1][j] = wmma_bf16(a1, bb, acc[1][j]);
        }
    }

    const int l = threadIdx.x & 31;
    const int col = l & 15;
    const int rb = (l >> 4) * 8;
#pragma unroll
    for (int i = 0; i < 2; ++i) {
#pragma unroll
        for (int j = 0; j < 4; ++j) {
#pragma unroll
            for (int v = 0; v < 8; ++v) {
                const int mm = m0 + i * 16 + rb + v;
                const int nn = n0 + j * 16 + col;
                float x = (acc[i][j][v] + bias[nn]) * scale;
                if (MODE == 0) {
                    ((bf16*)outp)[(size_t)mm * N + nn] = (bf16)x;
                } else if (MODE == 1) {
                    const int bidx = mm / rows_per_batch;
                    const int mr = mm % rows_per_batch;
                    ((bf16*)outp)[(long long)bidx * out_batch_stride +
                                  (size_t)nn * rows_per_batch + mr] = (bf16)x;
                } else {
                    ((float*)outp)[(size_t)mm * N + nn] = x;
                }
            }
        }
    }
}

// ---------------------------------------------------------------------------
// Flash attention. One wave per 32-query tile; the 4 waves of a block share
// one (b,h) and, when async-LDS is available, cooperatively stage each 64-key
// K/V block into LDS via GLOBAL_LOAD_ASYNC_TO_LDS_B128 (ASYNCcnt + barrier).
// ---------------------------------------------------------------------------
__global__ __launch_bounds__(128) void flash_attn_kernel(
    const bf16* __restrict__ Q, const bf16* __restrict__ Kb,
    const bf16* __restrict__ Vt, const uint32_t* __restrict__ maskp,
    bf16* __restrict__ O) {
    __shared__ __align__(16) bf16 plds[4][MT_][16][72];
#ifdef HAVE_ASYNC_LDS
    __shared__ __align__(16) bf16 klds[64][HD_ + PAD_];  // [key][d]
    __shared__ __align__(16) bf16 vlds[HD_][64 + PAD_];  // [d][key]
#endif

    const int l = threadIdx.x & 31;
    const int widx = threadIdx.x >> 5;
    const int w = blockIdx.x * 4 + widx;
    const int tiles_per_head = NQ_ / (16 * MT_);  // 32 (divisible by 4)
    const int b = w / (H_ * tiles_per_head);
    const int rem = w % (H_ * tiles_per_head);
    const int h = rem / tiles_per_head;
    const int m0 = (rem % tiles_per_head) * (16 * MT_);

    const bf16* Qp = Q + ((size_t)b * NQ_ + m0) * QD_ + h * HD_;
    const bf16* Kp = Kb + (size_t)b * NM_ * QD_ + h * HD_;
    const bf16* Vp = Vt + (size_t)b * QD_ * NM_ + (size_t)h * HD_ * NM_;
    const uint32_t* Mpk = maskp + ((size_t)b * NQ_ + m0) * (NM_ / 32);

    v16bf qa[MT_][2];
#pragma unroll
    for (int mt = 0; mt < MT_; ++mt) {
        qa[mt][0] = load_a_frag(Qp + (size_t)mt * 16 * QD_, QD_);
        qa[mt][1] = load_a_frag(Qp + (size_t)mt * 16 * QD_ + 32, QD_);
    }

    v8f o[MT_][4];
    float mrow[MT_][8], lrow[MT_][8];
#pragma unroll
    for (int mt = 0; mt < MT_; ++mt) {
#pragma unroll
        for (int j = 0; j < 4; ++j) o[mt][j] = {};
#pragma unroll
        for (int v = 0; v < 8; ++v) { mrow[mt][v] = -1e30f; lrow[mt][v] = 0.f; }
    }

    const int col = l & 15;
    const int rb = (l >> 4) * 8;
#ifdef HAVE_ASYNC_LDS
    const int srow = threadIdx.x >> 1;          // 0..63 (key / d index)
    const int spart = (threadIdx.x & 1) * 32;   // element offset 0 or 32
#endif

    for (int n0 = 0; n0 < NM_; n0 += 64) {
#ifdef HAVE_ASYNC_LDS
        // ---- cooperative async stage of K(64x64) and V^T(64x64) ---------
#pragma unroll
        for (int i = 0; i < 4; ++i)
            async_copy16(Kp + (size_t)(n0 + srow) * QD_ + spart + i * 8,
                         &klds[srow][spart + i * 8]);
#pragma unroll
        for (int i = 0; i < 4; ++i)
            async_copy16(Vp + (size_t)srow * NM_ + n0 + spart + i * 8,
                         &vlds[srow][spart + i * 8]);
        async_wait_all();
        __syncthreads();
#else
        if (n0 + 64 < NM_)
            __builtin_prefetch(Kp + (size_t)(n0 + 64) * QD_, 0, 1);
#endif

        // ---- S = Q @ K^T (pre-scaled, log2 domain) -----------------------
        v8f st[MT_][4];
#pragma unroll
        for (int mt = 0; mt < MT_; ++mt)
#pragma unroll
            for (int nt = 0; nt < 4; ++nt) st[mt][nt] = {};
#pragma unroll
        for (int nt = 0; nt < 4; ++nt)
#pragma unroll
            for (int kk = 0; kk < 2; ++kk) {
#ifdef HAVE_ASYNC_LDS
                v16bf bk = load_b_frag(&klds[nt * 16][kk * 32], HD_ + PAD_);
#else
                v16bf bk = load_b_frag(
                    Kp + (size_t)(n0 + nt * 16) * QD_ + kk * 32, QD_);
#endif
#pragma unroll
                for (int mt = 0; mt < MT_; ++mt)
                    st[mt][nt] = wmma_bf16(qa[mt][kk], bk, st[mt][nt]);
            }

        // ---- bit-packed mask: one aligned u64 per row per 64-key block ---
#pragma unroll
        for (int mt = 0; mt < MT_; ++mt)
#pragma unroll
            for (int v = 0; v < 8; ++v) {
                const uint64_t mw = *reinterpret_cast<const uint64_t*>(
                    Mpk + (size_t)(mt * 16 + rb + v) * (NM_ / 32) + (n0 >> 5));
#pragma unroll
                for (int nt = 0; nt < 4; ++nt)
                    if (!((mw >> (nt * 16 + col)) & 1ull))
                        st[mt][nt][v] = -1e30f;
            }

        // ---- online softmax (base 2) ------------------------------------
#pragma unroll
        for (int mt = 0; mt < MT_; ++mt) {
            float mnew[8], alpha[8];
#pragma unroll
            for (int v = 0; v < 8; ++v) {
                float rm = fmaxf(fmaxf(st[mt][0][v], st[mt][1][v]),
                                 fmaxf(st[mt][2][v], st[mt][3][v]));
                rm = half_max(rm);
                mnew[v] = fmaxf(mrow[mt][v], rm);
                alpha[v] = exp2f(mrow[mt][v] - mnew[v]);
                mrow[mt][v] = mnew[v];
            }
#pragma unroll
            for (int nt = 0; nt < 4; ++nt)
#pragma unroll
                for (int v = 0; v < 8; ++v)
                    st[mt][nt][v] = exp2f(st[mt][nt][v] - mnew[v]);
#pragma unroll
            for (int v = 0; v < 8; ++v) {
                float rs = st[mt][0][v] + st[mt][1][v] + st[mt][2][v] +
                           st[mt][3][v];
                rs = half_sum(rs);
                lrow[mt][v] = lrow[mt][v] * alpha[v] + rs;
            }
#pragma unroll
            for (int nt = 0; nt < 4; ++nt)
#pragma unroll
                for (int v = 0; v < 8; ++v) o[mt][nt][v] *= alpha[v];

            // P (C/D layout) -> LDS
#pragma unroll
            for (int nt = 0; nt < 4; ++nt)
#pragma unroll
                for (int v = 0; v < 8; ++v)
                    plds[widx][mt][rb + v][nt * 16 + col] =
                        (bf16)st[mt][nt][v];
        }
        asm volatile("s_wait_dscnt 0x0" ::: "memory");

        v16bf pa[MT_][2];
#pragma unroll
        for (int mt = 0; mt < MT_; ++mt) {
            pa[mt][0] = load_a_frag(&plds[widx][mt][0][0], 72);
            pa[mt][1] = load_a_frag(&plds[widx][mt][0][0] + 32, 72);
        }

        // ---- O += P @ V --------------------------------------------------
#pragma unroll
        for (int nt = 0; nt < 4; ++nt)
#pragma unroll
            for (int kk = 0; kk < 2; ++kk) {
#ifdef HAVE_ASYNC_LDS
                v16bf bv = load_b_frag(&vlds[nt * 16][kk * 32], 64 + PAD_);
#else
                v16bf bv = load_b_frag(
                    Vp + (size_t)(nt * 16) * NM_ + n0 + kk * 32, NM_);
#endif
#pragma unroll
                for (int mt = 0; mt < MT_; ++mt)
                    o[mt][nt] = wmma_bf16(pa[mt][kk], bv, o[mt][nt]);
            }

#ifdef HAVE_ASYNC_LDS
        __syncthreads();  // protect klds/vlds against next iteration (WAR)
#endif
    }

    // ---- normalize + store ----------------------------------------------
    bf16* Op = O + ((size_t)b * NQ_ + m0) * QD_ + h * HD_;
#pragma unroll
    for (int mt = 0; mt < MT_; ++mt)
#pragma unroll
        for (int v = 0; v < 8; ++v) {
            const float inv =
                lrow[mt][v] > 0.f ? 1.0f / lrow[mt][v] : 0.f;
#pragma unroll
            for (int nt = 0; nt < 4; ++nt)
                Op[(size_t)(mt * 16 + rb + v) * QD_ + nt * 16 + col] =
                    (bf16)(o[mt][nt][v] * inv);
        }
}

// ---------------------------------------------------------------------------
// Host-side orchestration
// ---------------------------------------------------------------------------
extern "C" void kernel_launch(void* const* d_in, const int* in_sizes, int n_in,
                              void* d_out, int out_size, void* d_ws,
                              size_t ws_size, hipStream_t stream) {
    (void)in_sizes; (void)n_in; (void)out_size; (void)ws_size;

    const float* query  = (const float*)d_in[0];
    const float* memory = (const float*)d_in[1];
    const int*   amask  = (const int*)d_in[2];
    const float* Wq = (const float*)d_in[3];
    const float* bq = (const float*)d_in[4];
    const float* Wk = (const float*)d_in[5];
    const float* bk = (const float*)d_in[6];
    const float* Wv = (const float*)d_in[7];
    const float* bv = (const float*)d_in[8];
    const float* Wo = (const float*)d_in[9];
    const float* bo = (const float*)d_in[10];

    const long long NQel = (long long)B_ * NQ_ * QD_;
    const long long NMel = (long long)B_ * NM_ * QD_;
    const long long Wel  = (long long)QD_ * QD_;
    const long long Mel  = (long long)B_ * NQ_ * NM_;

    char* ws = (char*)d_ws;
    bf16* qbf  = (bf16*)(ws);                        // 8 MB
    bf16* mbf  = (bf16*)(ws + (8LL << 20));          // 32 MB
    bf16* wqbf = (bf16*)(ws + (40LL << 20));         // 2 MB
    bf16* wkbf = (bf16*)(ws + (42LL << 20));         // 2 MB
    bf16* wvbf = (bf16*)(ws + (44LL << 20));         // 2 MB
    bf16* wobf = (bf16*)(ws + (46LL << 20));         // 2 MB
    bf16* Qbuf = (bf16*)(ws + (48LL << 20));         // 8 MB (scaled q)
    bf16* Kbuf = (bf16*)(ws + (56LL << 20));         // 32 MB
    bf16* Vtbf = (bf16*)(ws + (88LL << 20));         // 32 MB (per-batch V^T)
    bf16* Obuf = (bf16*)(ws + (120LL << 20));        // 8 MB
    uint32_t* mpk = (uint32_t*)(ws + (128LL << 20)); // 2 MB bit-packed mask

    cvt_f32_bf16<<<1024, 256, 0, stream>>>(query,  qbf,  NQel);
    cvt_f32_bf16<<<1024, 256, 0, stream>>>(memory, mbf,  NMel);
    cvt_f32_bf16<<<256,  256, 0, stream>>>(Wq, wqbf, Wel);
    cvt_f32_bf16<<<256,  256, 0, stream>>>(Wk, wkbf, Wel);
    cvt_f32_bf16<<<256,  256, 0, stream>>>(Wv, wvbf, Wel);
    cvt_f32_bf16<<<256,  256, 0, stream>>>(Wo, wobf, Wel);
    pack_mask_kernel<<<4096, 256, 0, stream>>>(amask, mpk, Mel);

    {   // Q = (query @ Wq^T + bq) * SCALE * log2(e)
        const int M = B_ * NQ_, N = QD_, K = QD_;
        const int waves = (M / 32) * (N / 64);
        gemm_bf16_kernel<0><<<waves / 8, 256, 0, stream>>>(
            qbf, wqbf, bq, Qbuf, M, N, K, SCALE_ * LOG2E_, 0, 0);
    }
    {   // K = memory @ Wk^T + bk
        const int M = B_ * NM_, N = QD_, K = QD_;
        const int waves = (M / 32) * (N / 64);
        gemm_bf16_kernel<0><<<waves / 8, 256, 0, stream>>>(
            mbf, wkbf, bk, Kbuf, M, N, K, 1.0f, 0, 0);
    }
    {   // V^T (per-batch transposed)
        const int M = B_ * NM_, N = QD_, K = QD_;
        const int waves = (M / 32) * (N / 64);
        gemm_bf16_kernel<1><<<waves / 8, 256, 0, stream>>>(
            mbf, wvbf, bv, Vtbf, M, N, K, 1.0f, NM_,
            (long long)QD_ * NM_);
    }
    {   // fused masked softmax attention
        const int total_waves = B_ * H_ * (NQ_ / (16 * MT_));  // 2048
        flash_attn_kernel<<<total_waves / 4, 128, 0, stream>>>(
            Qbuf, Kbuf, Vtbf, mpk, Obuf);
    }
    {   // out = Obuf @ Wo^T + bo -> fp32
        const int M = B_ * NQ_, N = QD_, K = QD_;
        const int waves = (M / 32) * (N / 64);
        gemm_bf16_kernel<2><<<waves / 8, 256, 0, stream>>>(
            Obuf, wobf, bo, d_out, M, N, K, 1.0f, 0, 0);
    }
}
